// NodeModel_18837726561015
// MI455X (gfx1250) — matrix operand order repaired and checked by reading, not compiled, not run on hardware
//
#include <hip/hip_runtime.h>

typedef __attribute__((ext_vector_type(16))) _Float16 v16h;
typedef __attribute__((ext_vector_type(8)))  _Float16 v8h;
typedef __attribute__((ext_vector_type(8)))  float    v8f;

#define N_NODES 50000
#define N_EDGES 800000

// D = A*B + C, 16x16x32 f16 -> f32
__device__ __forceinline__ v8f wmma_f16(v16h a, v16h b, v8f c) {
  return __builtin_amdgcn_wmma_f32_16x16x32_f16(
      /*neg_a=*/false, a, /*neg_b=*/false, b,
      /*c_mod=*/(short)0, c, /*reuse_a=*/false, /*reuse_b=*/false);
}

// Stage a row-major f32 weight [K][N] into LDS pre-swizzled so that each
// B fragment (kc, nt) is 32 lanes x 16 contiguous f16 (32B per lane).
// B[k][n] -> lane = (((k%32)/8)%2)*16 + n%16, slot = (k%8) + ((k%32)/16)*8.
__device__ __forceinline__ void stage_b_swizzled(const float* __restrict__ W,
                                                 _Float16* sW, int K, int N,
                                                 int tid, int nthreads) {
  const int NT = N >> 4;
  for (int i = tid; i < K * N; i += nthreads) {
    const int k = i / N, n = i % N;
    const int kc = k >> 5, klocal = k & 31;
    const int khb = (klocal >> 3) & 1, klr = (klocal & 7) | ((klocal >> 4) << 3);
    const int nt = n >> 4, nl = n & 15;
    const int frag = kc * NT + nt;
    sW[(frag * 32 + khb * 16 + nl) * 16 + klr] = (_Float16)W[i];
  }
}

// One contiguous 32-byte per-lane read -> full B fragment.
__device__ __forceinline__ v16h load_b_frag(const _Float16* sW, int NT, int kc,
                                            int nt, int lane) {
  return *(const v16h*)(sW + ((size_t)(kc * NT + nt) * 32 + lane) * 16);
}

// Convert 8 consecutive f32 (two float4 loads) into A-fragment elements
// [e0..e0+7] (the f16 A layout's per-lane K indices are two runs of 8).
__device__ __forceinline__ void fill_a_run(v16h& a, int e0, const float* p) {
  const float4 q0 = *(const float4*)(p);
  const float4 q1 = *(const float4*)(p + 4);
  a[e0 + 0] = (_Float16)q0.x; a[e0 + 1] = (_Float16)q0.y;
  a[e0 + 2] = (_Float16)q0.z; a[e0 + 3] = (_Float16)q0.w;
  a[e0 + 4] = (_Float16)q1.x; a[e0 + 5] = (_Float16)q1.y;
  a[e0 + 6] = (_Float16)q1.z; a[e0 + 7] = (_Float16)q1.w;
}

// Layer-2 A fragment from a 32-f16 row-major hidden row in LDS:
// elements 0..7  <- k = 8*khalf + 0..7
// elements 8..15 <- k = 16 + 8*khalf + 0..7
__device__ __forceinline__ v16h load_a_frag_lds(const _Float16* hrow, int khalf) {
  const v8h lo = *(const v8h*)(hrow + 8 * khalf);
  const v8h hi = *(const v8h*)(hrow + 16 + 8 * khalf);
  return __builtin_shufflevector(lo, hi, 0, 1, 2, 3, 4, 5, 6, 7,
                                 8, 9, 10, 11, 12, 13, 14, 15);
}

// --------------------------------------------------------------------------
// Kernel 0: zero workspace (agg_sum, agg_max bits, cnt)
// --------------------------------------------------------------------------
__global__ void zero_ws_kernel(int* __restrict__ p, size_t n) {
  size_t i = (size_t)blockIdx.x * blockDim.x + threadIdx.x;
  const size_t stride = (size_t)gridDim.x * blockDim.x;
  for (; i < n; i += stride) p[i] = 0;
}

// --------------------------------------------------------------------------
// Kernel 1: edge MLP + scatter (sum, max-bits, count).  16 edges per wave.
// --------------------------------------------------------------------------
__global__ __launch_bounds__(256) void edge_msg_kernel(
    const float* __restrict__ x, const int* __restrict__ ei,
    const float* __restrict__ ea,
    const float* __restrict__ W1, const float* __restrict__ b1,
    const float* __restrict__ W2, const float* __restrict__ b2,
    float* __restrict__ agg_sum, int* __restrict__ agg_max,
    float* __restrict__ cnt) {
  __shared__ __align__(32) _Float16 sW1[192 * 32];  // swizzled, 6 kc x 2 nt
  __shared__ __align__(32) _Float16 sW2[32 * 64];   // swizzled, 1 kc x 4 nt
  __shared__ __align__(32) _Float16 sH[8][16 * 32];

  const int tid = threadIdx.x;
  stage_b_swizzled(W1, sW1, 192, 32, tid, 256);
  stage_b_swizzled(W2, sW2, 32, 64, tid, 256);
  __syncthreads();

  const int wave  = tid >> 5;
  const int lane  = tid & 31;
  const int l16   = lane & 15;
  const int khalf = lane >> 4;

  const int tile = blockIdx.x * 8 + wave;  // 800000/16 = 50000 tiles, exact
  const int em = tile * 16 + l16;          // this lane's A-matrix row (edge)
  const int er = ei[em];                   // source node
  const int ec = ei[N_EDGES + em];         // destination node

  // ---- layer 1: [16,192] @ [192,32], K in 6 chunks of 32, N in 2 tiles ----
  v8f acc0 = {}, acc1 = {};
#pragma unroll
  for (int kc = 0; kc < 6; ++kc) {
    const float* src;
    int fo;
    if (kc < 2)      { src = x + (size_t)er * 64;  fo = kc * 32; }
    else if (kc < 4) { src = x + (size_t)ec * 64;  fo = kc * 32 - 64; }
    else             { src = ea + (size_t)em * 64; fo = kc * 32 - 128; }
    v16h a;
    fill_a_run(a, 0, src + fo + 8 * khalf);
    fill_a_run(a, 8, src + fo + 16 + 8 * khalf);
    acc0 = wmma_f16(a, load_b_frag(sW1, 2, kc, 0, lane), acc0);
    acc1 = wmma_f16(a, load_b_frag(sW1, 2, kc, 1, lane), acc1);
  }

  // bias + relu -> hidden row-major f16 (layer-2 A source)
  {
    const float bv0 = b1[l16], bv1 = b1[16 + l16];
#pragma unroll
    for (int r = 0; r < 8; ++r) {
      const int m = r + 8 * khalf;
      sH[wave][m * 32 + l16]      = (_Float16)fmaxf(acc0[r] + bv0, 0.0f);
      sH[wave][m * 32 + 16 + l16] = (_Float16)fmaxf(acc1[r] + bv1, 0.0f);
    }
  }
  __syncthreads();

  // ---- layer 2: [16,32] @ [32,64] ----
  const v16h a2 = load_a_frag_lds(&sH[wave][l16 * 32], khalf);

#pragma unroll
  for (int nt = 0; nt < 4; ++nt) {
    v8f acc = {};
    acc = wmma_f16(a2, load_b_frag(sW2, 4, 0, nt, lane), acc);
    const int fcol = nt * 16 + l16;
    const float bv = b2[fcol];
#pragma unroll
    for (int r = 0; r < 8; ++r) {
      const float val = fmaxf(acc[r] + bv, 0.0f);     // >= 0 -> int-ordered
      const int dst = __shfl(ec, r + 8 * khalf, 32);  // dest node of row m
      atomicAdd(agg_sum + (size_t)dst * 64 + fcol, val);
      atomicMax(agg_max + (size_t)dst * 64 + fcol, __float_as_int(val));
    }
  }
  if (khalf == 0) atomicAdd(cnt + ec, 1.0f);  // one per edge
}

// --------------------------------------------------------------------------
// Kernel 2: node MLP.  16 nodes per wave, 5 waves per block.
// --------------------------------------------------------------------------
__global__ __launch_bounds__(160) void node_mlp_kernel(
    const float* __restrict__ x, const float* __restrict__ u,
    const int* __restrict__ batch,
    const float* __restrict__ W3, const float* __restrict__ b3,
    const float* __restrict__ W4, const float* __restrict__ b4,
    const float* __restrict__ agg_sum, const int* __restrict__ agg_max,
    const float* __restrict__ cnt, float* __restrict__ out) {
  __shared__ __align__(32) _Float16 sW3[256 * 32];  // swizzled, 8 kc x 2 nt
  __shared__ __align__(32) _Float16 sW4[32 * 128];  // swizzled, 1 kc x 8 nt
  __shared__ __align__(32) _Float16 sH[5][16 * 32];

  const int tid = threadIdx.x;
  stage_b_swizzled(W3, sW3, 256, 32, tid, 160);
  stage_b_swizzled(W4, sW4, 32, 128, tid, 160);
  __syncthreads();

  const int wave  = tid >> 5;
  const int lane  = tid & 31;
  const int l16   = lane & 15;
  const int khalf = lane >> 4;

  const int tile = blockIdx.x * 5 + wave;  // 50000/16 = 3125 tiles, exact
  const int node = tile * 16 + l16;        // this lane's A-matrix row (node)
  const int g = batch[node];
  const float inv = 1.0f / fmaxf(cnt[node], 1.0f);

  // ---- layer 1: [16,256] @ [256,32], 8 K-chunks, 2 N-tiles ----
  v8f acc0 = {}, acc1 = {};
#pragma unroll
  for (int kc = 0; kc < 8; ++kc) {
    v16h a;
#pragma unroll
    for (int h2 = 0; h2 < 2; ++h2) {
      const int f = kc * 32 + h2 * 16 + 8 * khalf;  // region uniform per kc
      if (kc < 2) {
        fill_a_run(a, h2 * 8, x + (size_t)node * 64 + f);
      } else if (kc < 4) {
        const float4 q0 = *(const float4*)(agg_sum + (size_t)node * 64 + (f - 64));
        const float4 q1 = *(const float4*)(agg_sum + (size_t)node * 64 + (f - 64) + 4);
        a[h2 * 8 + 0] = (_Float16)(q0.x * inv); a[h2 * 8 + 1] = (_Float16)(q0.y * inv);
        a[h2 * 8 + 2] = (_Float16)(q0.z * inv); a[h2 * 8 + 3] = (_Float16)(q0.w * inv);
        a[h2 * 8 + 4] = (_Float16)(q1.x * inv); a[h2 * 8 + 5] = (_Float16)(q1.y * inv);
        a[h2 * 8 + 6] = (_Float16)(q1.z * inv); a[h2 * 8 + 7] = (_Float16)(q1.w * inv);
      } else if (kc < 6) {
        const int4 q0 = *(const int4*)(agg_max + (size_t)node * 64 + (f - 128));
        const int4 q1 = *(const int4*)(agg_max + (size_t)node * 64 + (f - 128) + 4);
        a[h2 * 8 + 0] = (_Float16)__int_as_float(q0.x);
        a[h2 * 8 + 1] = (_Float16)__int_as_float(q0.y);
        a[h2 * 8 + 2] = (_Float16)__int_as_float(q0.z);
        a[h2 * 8 + 3] = (_Float16)__int_as_float(q0.w);
        a[h2 * 8 + 4] = (_Float16)__int_as_float(q1.x);
        a[h2 * 8 + 5] = (_Float16)__int_as_float(q1.y);
        a[h2 * 8 + 6] = (_Float16)__int_as_float(q1.z);
        a[h2 * 8 + 7] = (_Float16)__int_as_float(q1.w);
      } else {
        fill_a_run(a, h2 * 8, u + (size_t)g * 64 + (f - 192));
      }
    }
    acc0 = wmma_f16(a, load_b_frag(sW3, 2, kc, 0, lane), acc0);
    acc1 = wmma_f16(a, load_b_frag(sW3, 2, kc, 1, lane), acc1);
  }

  {
    const float bv0 = b3[l16], bv1 = b3[16 + l16];
#pragma unroll
    for (int r = 0; r < 8; ++r) {
      const int m = r + 8 * khalf;
      sH[wave][m * 32 + l16]      = (_Float16)fmaxf(acc0[r] + bv0, 0.0f);
      sH[wave][m * 32 + 16 + l16] = (_Float16)fmaxf(acc1[r] + bv1, 0.0f);
    }
  }
  __syncthreads();

  // ---- layer 2: [16,32] @ [32,128], 8 N-tiles ----
  const v16h a2 = load_a_frag_lds(&sH[wave][l16 * 32], khalf);

#pragma unroll
  for (int nt = 0; nt < 8; ++nt) {
    v8f acc = {};
    acc = wmma_f16(a2, load_b_frag(sW4, 8, 0, nt, lane), acc);
    const int fcol = nt * 16 + l16;
    const float bv = b4[fcol];
#pragma unroll
    for (int r = 0; r < 8; ++r) {
      const int outrow = tile * 16 + r + 8 * khalf;
      out[(size_t)outrow * 128 + fcol] = fmaxf(acc[r] + bv, 0.0f);
    }
  }
}

// --------------------------------------------------------------------------
extern "C" void kernel_launch(void* const* d_in, const int* in_sizes, int n_in,
                              void* d_out, int out_size, void* d_ws, size_t ws_size,
                              hipStream_t stream) {
  const float* x     = (const float*)d_in[0];
  const int*   ei    = (const int*)d_in[1];
  const float* ea    = (const float*)d_in[2];
  const float* u     = (const float*)d_in[3];
  const int*   batch = (const int*)d_in[4];
  const float* W1 = (const float*)d_in[5];
  const float* b1 = (const float*)d_in[6];
  const float* W2 = (const float*)d_in[7];
  const float* b2 = (const float*)d_in[8];
  const float* W3 = (const float*)d_in[9];
  const float* b3 = (const float*)d_in[10];
  const float* W4 = (const float*)d_in[11];
  const float* b4 = (const float*)d_in[12];
  float* out = (float*)d_out;

  // workspace: agg_sum [N,64] f32 | agg_max [N,64] float-bits | cnt [N] f32
  float* agg_sum = (float*)d_ws;
  int*   agg_max = (int*)((char*)d_ws + (size_t)N_NODES * 64 * sizeof(float));
  float* cnt     = (float*)((char*)d_ws + (size_t)N_NODES * 64 * 2 * sizeof(float));

  const size_t zwords = (size_t)N_NODES * 64 * 2 + (size_t)N_NODES;
  zero_ws_kernel<<<4096, 256, 0, stream>>>((int*)d_ws, zwords);

  // 800000 edges / 16 per wave / 8 waves per block = 6250 blocks (exact)
  edge_msg_kernel<<<N_EDGES / (16 * 8), 256, 0, stream>>>(
      x, ei, ea, W1, b1, W2, b2, agg_sum, agg_max, cnt);

  // 50000 nodes / 16 per wave / 5 waves per block = 625 blocks (exact)
  node_mlp_kernel<<<N_NODES / (16 * 5), 160, 0, stream>>>(
      x, u, batch, W3, b3, W4, b4, agg_sum, agg_max, cnt, out);
}
